// DecoderCRF_1331439862078
// MI455X (gfx1250) — matrix operand order repaired and checked by reading, not compiled, not run on hardware
//
#include <hip/hip_runtime.h>
#include <hip/hip_bf16.h>

// Problem sizes (fixed by reference)
constexpr int L_ = 2048;   // sequence length
constexpr int B_ = 64;     // batch
constexpr int H_ = 512;    // hidden
constexpr int T_ = 2;      // tags

// Kernel configuration
constexpr int NWG     = 16;          // persistent workgroups (one per WGP)
constexpr int THREADS = 256;         // 8 wave32 per WG
constexpr int HS      = H_ / NWG;    // 32 hidden units per WG
constexpr int NS      = 4 * HS;      // 128 gate rows per WG (i,f,g,o)
constexpr int KC      = H_ / 32;     // 16 K-chunks of 32
constexpr int GLP     = NS + 4;      // padded gate-buffer row (floats)
constexpr int ROWB    = H_ * 2;      // bytes per (unpadded) f16 row = 1024

// XOR-swizzled row layout: 16B chunk c of row r lives at chunk (c ^ (r & 15)).
// Row base (multiple of 1024B), chunk offset (<1024B) combine via XOR/OR freely.

// LDS layout byte offsets (dynamic LDS starts at offset 0 of the wave's allocation)
constexpr int WL_OFF  = 0;                                   // [NS][H] f16, swizzled
constexpr int HL_OFF  = WL_OFF + NS * ROWB;                  // [B_][H] f16, swizzled
constexpr int GL_OFF  = HL_OFF + B_ * ROWB;                  // [B_][GLP] f32
constexpr int CL_OFF  = GL_OFF + B_ * GLP * 4;               // [B_][HS] f32
constexpr int XS_OFF  = CL_OFF + B_ * HS * 4;                // [B_] f32
constexpr int WIH_OFF = XS_OFF + B_ * 4;                     // [NS] f32
constexpr int BS_OFF  = WIH_OFF + NS * 4;                    // [NS] f32
constexpr int WT_OFF  = BS_OFF + NS * 4;                     // [T_][HS] f32
constexpr int SMEM_BYTES = WT_OFF + T_ * HS * 4;

typedef __attribute__((ext_vector_type(16))) _Float16 v16h;
typedef __attribute__((ext_vector_type(8)))  float    v8f;
typedef __attribute__((ext_vector_type(4)))  unsigned v4u;

union HFrag { v16h v; v4u q[2]; };        // 16-bit A/B fragment: 2 x 16B chunks
union FFrag { v8f  v; float f[8]; };      // f32 C/D fragment

__device__ __forceinline__ float sigm(float x) { return 1.0f / (1.0f + __expf(-x)); }

__device__ __forceinline__ float logadd2(float a, float b) {
    float m = fmaxf(a, b);
    return m + __logf(__expf(a - m) + __expf(b - m));
}

// Sense-reversal grid barrier over NWG resident workgroups (ctrl[0]=count, ctrl[1]=generation)
__device__ __forceinline__ void grid_sync(unsigned* ctrl) {
    __syncthreads();
    if (threadIdx.x == 0) {
        __threadfence();                      // release
        volatile unsigned* vgen = (volatile unsigned*)(ctrl + 1);
        unsigned g = *vgen;
        if (atomicAdd(ctrl, 1u) == (unsigned)NWG - 1u) {
            ctrl[0] = 0u;
            __threadfence();
            atomicAdd(ctrl + 1, 1u);
        } else {
            while (*vgen == g) { __builtin_amdgcn_s_sleep(2); }
        }
        __threadfence();                      // acquire
    }
    __syncthreads();
}

// ---------------------------------------------------------------------------
__global__ void init_kernel(const float* __restrict__ hidden,
                            float* __restrict__ logits,
                            _Float16* __restrict__ hbuf,
                            unsigned* __restrict__ ctrl) {
    int i = blockIdx.x * blockDim.x + threadIdx.x;
    if (i < 2)            ctrl[i]   = 0u;
    if (i < L_ * B_ * T_) logits[i] = 0.0f;
    if (i < B_ * H_) {
        hbuf[i]           = (_Float16)hidden[i];
        hbuf[B_ * H_ + i] = (_Float16)0.0f;
    }
}

// ---------------------------------------------------------------------------
// Persistent LSTM kernel
// ---------------------------------------------------------------------------
__global__ __launch_bounds__(THREADS, 1)
void lstm_persistent_kernel(const float* __restrict__ xin,   // (L,B,1)
                            const float* __restrict__ Whh,   // (4H,H)
                            const float* __restrict__ Wih,   // (4H,1)
                            const float* __restrict__ bih,   // (4H)
                            const float* __restrict__ bhh,   // (4H)
                            const float* __restrict__ Wtag,  // (T,H)
                            _Float16* __restrict__ hbuf,     // (2,B,H) f16 ping-pong
                            float* __restrict__ logits,      // (L,B,T)
                            unsigned* __restrict__ ctrl) {
    extern __shared__ char smem[];
    char*     wlB   = smem + WL_OFF;               // W_hh slice, swizzled f16
    char*     hlB   = smem + HL_OFF;               // h_t, swizzled f16
    _Float16* Wl    = (_Float16*)wlB;
    float*    gl    = (float*)(smem + GL_OFF);
    float*    cl    = (float*)(smem + CL_OFF);
    float*    xs    = (float*)(smem + XS_OFF);
    float*    wih   = (float*)(smem + WIH_OFF);
    float*    bsum  = (float*)(smem + BS_OFF);
    float*    wtagl = (float*)(smem + WT_OFF);

    const int wg   = blockIdx.x;
    const int tid  = threadIdx.x;
    const int lane = tid & 31;
    const int wv   = tid >> 5;
    const int ml   = lane & 15;
    const int hi   = lane >> 4;

    // wave tile ownership: 2 M-tiles x 2 N-tiles
    const int mt0 = (wv & 1) * 2;       // M-tiles {mt0, mt0+1}
    const int nt0 = (wv >> 1) * 2;      // N-tiles {nt0, nt0+1}

    // ---- one-time prologue: pack W_hh slice (f32->f16, swizzled) + params ----
    for (int i = tid; i < NS * H_; i += THREADS) {
        int r = i >> 9, k = i & (H_ - 1);
        int grow = (r >> 5) * H_ + wg * HS + (r & 31);     // gate-major global row
        int dk = (((k >> 3) ^ (r & 15)) << 3) | (k & 7);   // swizzled half index
        Wl[r * H_ + dk] = (_Float16)Whh[grow * H_ + k];
    }
    if (tid < NS) {
        int grow = (tid >> 5) * H_ + wg * HS + (tid & 31);
        wih[tid]  = Wih[grow];
        bsum[tid] = bih[grow] + bhh[grow];
    }
    if (tid < T_ * HS) {
        int k = tid >> 5;
        wtagl[tid] = Wtag[k * H_ + wg * HS + (tid & 31)];
    }
    for (int i = tid; i < B_ * HS; i += THREADS) cl[i] = 0.0f;

    // ---- loop-invariant lane constants: full address = base ^ (kc*64 [+32]) ----
    // A frag chunk offsets (bytes): (kc*64) ^ pA and (kc*64+32) ^ pA, pA = (hi^ml)<<4
    // B frag chunk offsets (bytes): (kc*64) ^ pB0 and (kc*64) ^ pB1
    const int pA  = (hi ^ ml) << 4;
    const int pB0 = (hi << 4) ^ (ml << 4);
    const int pB1 = (32 | (hi << 4)) ^ (ml << 4);
    int baseA[2], baseB0[2], baseB1[2];
    #pragma unroll
    for (int i = 0; i < 2; ++i) {
        baseA[i]  = ((mt0 + i) * 16 + ml) * ROWB ^ pA;     // into hl
        int nrow  = ((nt0 + i) * 16 + ml) * ROWB;          // into Wl
        baseB0[i] = nrow ^ pB0;
        baseB1[i] = nrow ^ pB1;
    }

    __syncthreads();

    // ---- time loop ----
    for (int t = 0; t < L_; ++t) {
        // stage h_t into swizzled LDS via async b128 DMA (ASYNCcnt path)
        {
            const v4u* hsrc = (const v4u*)(hbuf + (size_t)(t & 1) * B_ * H_);
            #pragma unroll
            for (int rep = 0; rep < (B_ * H_) / (8 * THREADS); ++rep) {   // 16 chunks/thread
                int i = rep * THREADS + tid;
                int r = i >> 6, c = i & 63;                                // 64 chunks/row
                unsigned ldsoff = (unsigned)(HL_OFF + r * ROWB + ((c ^ (r & 15)) << 4));
                asm volatile("global_load_async_to_lds_b128 %0, %1, off"
                             :: "v"(ldsoff), "v"(hsrc + i)
                             : "memory");
            }
            if (tid < B_) xs[tid] = xin[t * B_ + tid];
            if (tid == 0 && t + 1 < L_) __builtin_prefetch(xin + (t + 1) * B_, 0, 1);
            asm volatile("s_wait_asynccnt 0x0" ::: "memory");
        }
        __syncthreads();

        // ---- WMMA GEMM: g[b, n] = sum_k h[b,k] * W[n,k], software-pipelined ----
        FFrag acc[2][2];
        #pragma unroll
        for (int i = 0; i < 2; ++i)
            #pragma unroll
            for (int j = 0; j < 2; ++j)
                #pragma unroll
                for (int v = 0; v < 8; ++v) acc[i][j].f[v] = 0.0f;

        HFrag Af[2][2], Bf[2][2];   // [buffer][tile]

        auto loadA = [&](HFrag* f, int kc) {
            #pragma unroll
            for (int i = 0; i < 2; ++i) {
                f[i].q[0] = *(const v4u*)(hlB + (baseA[i] ^ (kc * 64)));
                f[i].q[1] = *(const v4u*)(hlB + (baseA[i] ^ (kc * 64 + 32)));
            }
        };
        auto loadB = [&](HFrag* f, int kc) {
            #pragma unroll
            for (int j = 0; j < 2; ++j) {
                f[j].q[0] = *(const v4u*)(wlB + (baseB0[j] ^ (kc * 64)));
                f[j].q[1] = *(const v4u*)(wlB + (baseB1[j] ^ (kc * 64)));
            }
        };

        loadA(Af[0], 0);
        loadB(Bf[0], 0);
        #pragma unroll
        for (int kc = 0; kc < KC; ++kc) {
            const int cur = kc & 1, nxt = cur ^ 1;
            if (kc + 1 < KC) {                 // prefetch next K-chunk
                loadA(Af[nxt], kc + 1);
                loadB(Bf[nxt], kc + 1);
            }
            #pragma unroll
            for (int i = 0; i < 2; ++i)
                #pragma unroll
                for (int j = 0; j < 2; ++j)
                    acc[i][j].v = __builtin_amdgcn_wmma_f32_16x16x32_f16(
                        false, Af[cur][i].v, false, Bf[cur][j].v,
                        (short)0, acc[i][j].v, false, false);
        }

        // dump accumulators: D layout lanes0-15 M=v, lanes16-31 M=v+8; N=ml
        #pragma unroll
        for (int i = 0; i < 2; ++i) {
            const int m0 = (mt0 + i) * 16 + hi * 8;
            #pragma unroll
            for (int j = 0; j < 2; ++j) {
                const int col = (nt0 + j) * 16 + ml;
                #pragma unroll
                for (int v = 0; v < 8; ++v)
                    gl[(m0 + v) * GLP + col] = acc[i][j].f[v];
            }
        }
        __syncthreads();

        // ---- elementwise gate math + c/h update + fused tag projection ----
        _Float16* hdst = hbuf + (size_t)((t + 1) & 1) * B_ * H_ + wg * HS;
        #pragma unroll
        for (int rep = 0; rep < (B_ * HS) / THREADS; ++rep) {
            const int idx = rep * THREADS + tid;
            const int b = idx >> 5, j = idx & 31;   // each wave handles one batch row
            const float xv = xs[b];
            const float gi = gl[b * GLP + j]          + bsum[j]          + xv * wih[j];
            const float gf = gl[b * GLP + HS + j]     + bsum[HS + j]     + xv * wih[HS + j];
            const float gg = gl[b * GLP + 2 * HS + j] + bsum[2 * HS + j] + xv * wih[2 * HS + j];
            const float go = gl[b * GLP + 3 * HS + j] + bsum[3 * HS + j] + xv * wih[3 * HS + j];
            const float c  = sigm(gf) * cl[b * HS + j] + sigm(gi) * tanhf(gg);
            const float hv = sigm(go) * tanhf(c);
            cl[b * HS + j] = c;
            hdst[b * H_ + j] = (_Float16)hv;

            // partial tag logits: reduce the 32-wide hidden slice across the wave
            float p0 = hv * wtagl[j];
            float p1 = hv * wtagl[HS + j];
            #pragma unroll
            for (int off = 16; off > 0; off >>= 1) {
                p0 += __shfl_xor(p0, off, 32);
                p1 += __shfl_xor(p1, off, 32);
            }
            if (j == 0) {
                atomicAdd(&logits[(t * B_ + b) * T_ + 0], p0);
                atomicAdd(&logits[(t * B_ + b) * T_ + 1], p1);
            }
        }

        grid_sync(ctrl);   // h_{t+1} fully published before anyone stages it
    }
}

// ---------------------------------------------------------------------------
// CRF kernel: one thread per batch element.
// ---------------------------------------------------------------------------
__global__ void crf_kernel(const float* __restrict__ logits,
                           const int* __restrict__ tags,            // (B,L)
                           const unsigned char* __restrict__ masks, // (L,B) bool
                           const float* __restrict__ btag,
                           const float* __restrict__ strt,
                           const float* __restrict__ endt,
                           const float* __restrict__ trans,         // (T,T)
                           float* __restrict__ out) {
    const int b = threadIdx.x;
    const float bt0 = btag[0], bt1 = btag[1];
    const float t00 = trans[0], t01 = trans[1], t10 = trans[2], t11 = trans[3];

    float score = 0.0f, a0 = 0.0f, a1 = 0.0f;
    int prev = 0, seqlen = 0;

    for (int t = 0; t < L_; ++t) {
        float l0 = logits[(t * B_ + b) * T_ + 0] + bt0;
        float l1 = logits[(t * B_ + b) * T_ + 1] + bt1;
        float mx = fmaxf(l0, l1);
        float e0 = __expf(l0 - mx), e1 = __expf(l1 - mx);
        float inv = 1.0f / (e0 + e1);
        float p0 = e0 * inv, p1 = e1 * inv;          // emissions = softmax probs
        int   tg = tags[b * L_ + t];
        bool  mk = masks[t * B_ + b] != 0;

        if (t == 0) {
            score = strt[tg] + (tg ? p1 : p0);
            a0 = strt[0] + p0;
            a1 = strt[1] + p1;
        } else {
            float mf = mk ? 1.0f : 0.0f;
            score += mf * (trans[prev * T_ + tg] + (tg ? p1 : p0));
            float n0 = p0 + logadd2(a0 + t00, a1 + t10);
            float n1 = p1 + logadd2(a0 + t01, a1 + t11);
            if (mk) { a0 = n0; a1 = n1; }
        }
        if (mk) ++seqlen;
        prev = tg;
    }

    int last = tags[b * L_ + (seqlen - 1)];
    float numer = score + endt[last];
    float den   = logadd2(a0 + endt[0], a1 + endt[1]);

    __shared__ float red[B_];
    red[b] = numer - den;
    __syncthreads();
    if (b == 0) {
        float s = 0.0f;
        for (int i = 0; i < B_; ++i) s += red[i];
        out[0] = -s;
    }
}

// ---------------------------------------------------------------------------
extern "C" void kernel_launch(void* const* d_in, const int* in_sizes, int n_in,
                              void* d_out, int out_size, void* d_ws, size_t ws_size,
                              hipStream_t stream) {
    (void)in_sizes; (void)n_in; (void)out_size; (void)ws_size;

    const float*         xin   = (const float*)d_in[0];
    const float*         hid   = (const float*)d_in[1];
    const int*           tags  = (const int*)d_in[2];
    const unsigned char* masks = (const unsigned char*)d_in[3];
    const float*         Wih   = (const float*)d_in[4];
    const float*         Whh   = (const float*)d_in[5];
    const float*         bih   = (const float*)d_in[6];
    const float*         bhh   = (const float*)d_in[7];
    const float*         Wtag  = (const float*)d_in[8];
    const float*         btag  = (const float*)d_in[9];
    const float*         strt  = (const float*)d_in[10];
    const float*         endt  = (const float*)d_in[11];
    const float*         trans = (const float*)d_in[12];

    // Workspace layout: [ctrl 256B][hbuf 2*B*H f16 = 128KB][logits L*B*T f32 = 1MB]
    char*     ws     = (char*)d_ws;
    unsigned* ctrl   = (unsigned*)ws;
    _Float16* hbuf   = (_Float16*)(ws + 256);
    float*    logits = (float*)(ws + 256 + (size_t)2 * B_ * H_ * sizeof(_Float16));

    const int initN = L_ * B_ * T_;   // covers logits, hbuf seed, ctrl
    init_kernel<<<dim3((initN + THREADS - 1) / THREADS), dim3(THREADS), 0, stream>>>(
        hid, logits, hbuf, ctrl);

    (void)hipFuncSetAttribute(reinterpret_cast<const void*>(lstm_persistent_kernel),
                              hipFuncAttributeMaxDynamicSharedMemorySize, SMEM_BYTES);
    lstm_persistent_kernel<<<dim3(NWG), dim3(THREADS), SMEM_BYTES, stream>>>(
        xin, Whh, Wih, bih, bhh, Wtag, hbuf, logits, ctrl);

    crf_kernel<<<dim3(1), dim3(B_), 0, stream>>>(
        logits, tags, masks, btag, strt, endt, trans, (float*)d_out);
}
